// BottomUpNet_32822140076801
// MI455X (gfx1250) — matrix operand order repaired and checked by reading, not compiled, not run on hardware
//
#include <hip/hip_runtime.h>

// ---------------------------------------------------------------------------
// BottomUpNet on MI455X (gfx1250): recurrent MLP scan, bf16 WMMA + f32 accum.
//   N=8192 rows, K=16 steps, hidden=1024, n_in=64.
//   Per block: 64 rows resident in LDS for all 16 steps (2 ping-pong bf16
//   buffers), 8 waves split the 64 output-column tiles, sigmoid head fused.
//   Tower tiles double-buffered in LDS via GLOBAL_LOAD_ASYNC_TO_LDS_B128.
// ---------------------------------------------------------------------------

typedef __attribute__((ext_vector_type(16))) __bf16    v16bf;
typedef __attribute__((ext_vector_type(8)))  float     v8f;
typedef __attribute__((ext_vector_type(4)))  unsigned  u32x4;

#define R_ROWS   64          // rows per block
#define STRIDE   1048        // LDS activation row stride (elements), 16B-aligned, bank-spread
#define TSTRIDE  72          // LDS tower row stride
#define NT       64          // 1024/16 column tiles
#define NH       1024
#define N_TOTAL  8192
#define K_STEPS  16
#define N_IN     64

__device__ __forceinline__ unsigned short f2bf(float f) {
  unsigned u = __float_as_uint(f);
  u += 0x7FFFu + ((u >> 16) & 1u);     // round-to-nearest-even
  return (unsigned short)(u >> 16);
}

union FragU { u32x4 q[2]; v16bf v; };

// A fragment (16x32 bf16): lane holds row=lane%16; elements 0-15 -> K kb..kb+15
// (kb already includes (lane/16)*16).
__device__ __forceinline__ v16bf load_a(const unsigned short* p) {
  FragU u;
  u.q[0] = *(const u32x4*)(p);
  u.q[1] = *(const u32x4*)(p + 8);     // contiguous 32 bytes
  return u.v;
}

// B fragment (32x16 bf16), packed contiguous per lane by the prep kernel.
__device__ __forceinline__ v16bf load_b(const unsigned short* p) {
  FragU u;
  u.q[0] = *(const u32x4*)(p);
  u.q[1] = *(const u32x4*)(p + 8);     // contiguous 32 bytes
  return u.v;
}

// Async prefetch of one 64x64 bf16 tower tile into LDS (padded stride 72).
// Each lane moves 16B per instruction; 2 instructions/wave x 8 waves = 8KB.
__device__ __forceinline__ void async_tower_load(const unsigned short* gsrc,
                                                 unsigned short* TtBuf, int tid) {
#pragma unroll
  for (int j = 0; j < 2; ++j) {
    int idx = tid + j * 256;                 // 0..511 16B chunks
    int r = idx >> 3, c = idx & 7;           // row, 16B-chunk within 128B row
    unsigned lds  = (unsigned)(size_t)TtBuf + (unsigned)(r * TSTRIDE * 2 + c * 16);
    unsigned goff = (unsigned)(r * N_IN * 2 + c * 16);
    asm volatile("global_load_async_to_lds_b128 %0, %1, %2"
                 :: "v"(lds), "v"(goff), "s"(gsrc) : "memory");
  }
}

__device__ __forceinline__ void wait_async0() {
  asm volatile("s_wait_asynccnt 0" ::: "memory");
}

// One fused layer: Y = relu(X(@tower) * W + b). If FUSE, instead accumulate
// relu(...)*OW3 into per-row pred partials (GEMM2 of the output head).
template <int KT, bool FUSE>
__device__ __forceinline__ void gemm_layer(
    const unsigned short* __restrict__ Xs,   // LDS activations (stride STRIDE)
    const unsigned short* __restrict__ Ts,   // LDS tower tile (stride TSTRIDE), kt>=32
    const unsigned short* __restrict__ Wp,   // packed bf16 weights (global/L2)
    const float*          __restrict__ biasS,// LDS bias (1024)
    unsigned short*       __restrict__ Ys,   // LDS output (unused if FUSE)
    const float*          __restrict__ OW3s, // LDS OW3 (if FUSE)
    float*                __restrict__ predp,// LDS pred partials (if FUSE)
    int lane, int wave)
{
  const int rlo  = lane & 15;
  const int hi   = lane >> 4;
  const int koff = hi * 16;

  float predacc[4][8];
  if (FUSE)
    for (int r = 0; r < 4; ++r)
      for (int v = 0; v < 8; ++v) predacc[r][v] = 0.f;

  const v8f vzero = {0.f,0.f,0.f,0.f,0.f,0.f,0.f,0.f};

  for (int ctb = 0; ctb < 4; ++ctb) {
    const int ct0 = wave * 8 + ctb * 2;
    v8f acc[4][2];
    for (int r = 0; r < 4; ++r) { acc[r][0] = vzero; acc[r][1] = vzero; }

#pragma unroll 2
    for (int kt = 0; kt < KT; ++kt) {
      const unsigned short* abase;
      int astride, kb;
      if (KT > 32 && kt >= 32) { abase = Ts; astride = TSTRIDE; kb = (kt - 32) * 32 + koff; }
      else                     { abase = Xs; astride = STRIDE;  kb = kt * 32 + koff; }

      v16bf A[4];
#pragma unroll
      for (int rt = 0; rt < 4; ++rt)
        A[rt] = load_a(abase + (rt * 16 + rlo) * astride + kb);

#pragma unroll
      for (int c2 = 0; c2 < 2; ++c2) {
        const unsigned short* bp = Wp + (size_t)(kt * NT + (ct0 + c2)) * 512 + lane * 16;
        v16bf B = load_b(bp);
#pragma unroll
        for (int rt = 0; rt < 4; ++rt)
          acc[rt][c2] = __builtin_amdgcn_wmma_f32_16x16x32_bf16(
              false, A[rt], false, B, (short)0, acc[rt][c2], false, false);
      }
    }

    // epilogue: C layout lane holds (row = rt*16 + v + 8*hi, col = ct*16 + lane%16)
#pragma unroll
    for (int c2 = 0; c2 < 2; ++c2) {
      const int col = (ct0 + c2) * 16 + rlo;
      const float bb = biasS[col];
      const float w3 = FUSE ? OW3s[col] : 0.f;
#pragma unroll
      for (int rt = 0; rt < 4; ++rt) {
#pragma unroll
        for (int v = 0; v < 8; ++v) {
          float h = fmaxf(acc[rt][c2][v] + bb, 0.f);
          if (FUSE) predacc[rt][v] += h * w3;
          else      Ys[(rt * 16 + v + 8 * hi) * STRIDE + col] = f2bf(h);
        }
      }
    }
  }

  if (FUSE) {
    // reduce over the 16 lanes (= columns) sharing the same rows, then one
    // ds_add_f32 per row for the cross-wave reduction.
#pragma unroll
    for (int rt = 0; rt < 4; ++rt) {
#pragma unroll
      for (int v = 0; v < 8; ++v) {
        float s = predacc[rt][v];
        s += __shfl_xor(s, 1);
        s += __shfl_xor(s, 2);
        s += __shfl_xor(s, 4);
        s += __shfl_xor(s, 8);
        if (rlo == 0) atomicAdd(&predp[rt * 16 + v + 8 * hi], s);
      }
    }
  }
}

__global__ __launch_bounds__(256, 1) void bottomup_main(
    const unsigned short* __restrict__ MW1p, const unsigned short* __restrict__ MW2p,
    const unsigned short* __restrict__ MW3p, const unsigned short* __restrict__ OW1p,
    const unsigned short* __restrict__ OW2p, const unsigned short* __restrict__ towersb,
    const float* __restrict__ aggregate,
    const float* __restrict__ Mb1, const float* __restrict__ Mb2, const float* __restrict__ Mb3,
    const float* __restrict__ Ob1, const float* __restrict__ Ob2,
    const float* __restrict__ OW3, const float* __restrict__ Ob3,
    float* __restrict__ out)
{
  extern __shared__ __align__(16) char smem[];
  unsigned short* bufA = (unsigned short*)smem;                 // 64 x 1048 bf16
  unsigned short* bufB = bufA + R_ROWS * STRIDE;                // 64 x 1048 bf16
  unsigned short* Tt0  = bufB + R_ROWS * STRIDE;                // 64 x 72   bf16
  unsigned short* Tt1  = Tt0  + R_ROWS * TSTRIDE;               // 64 x 72   bf16
  float* bOb1  = (float*)(Tt1 + R_ROWS * TSTRIDE);
  float* bOb2  = bOb1 + NH;
  float* bMb1  = bOb2 + NH;
  float* bMb2  = bMb1 + NH;
  float* bMb3  = bMb2 + NH;
  float* sOW3  = bMb3 + NH;
  float* predp = sOW3 + NH;                                     // 64 f32
  float* prod  = predp + R_ROWS;                                // 64 f32

  const int tid  = threadIdx.x;
  const int lane = tid & 31;
  const int wave = tid >> 5;
  const int row0 = blockIdx.x * R_ROWS;

  // kick off async prefetch of tower tile 0 while we stage biases / summary
  const unsigned short* tbase = towersb + (size_t)row0 * N_IN;
  async_tower_load(tbase, Tt0, tid);

  for (int i = tid; i < NH; i += 256) {
    bOb1[i] = Ob1[i]; bOb2[i] = Ob2[i];
    bMb1[i] = Mb1[i]; bMb2[i] = Mb2[i]; bMb3[i] = Mb3[i];
    sOW3[i] = OW3[i];
  }
  if (tid < R_ROWS) { predp[tid] = 0.f; prod[tid] = 1.f; }
  for (int i = tid; i < R_ROWS * NH; i += 256) {               // summary0 = aggregate
    int r = i >> 10, c = i & (NH - 1);
    bufA[r * STRIDE + c] = f2bf(aggregate[c]);
  }

  unsigned short* Sbuf = bufA;
  unsigned short* Hbuf = bufB;
  const float ob3 = Ob3[0];

  for (int k = 0; k < K_STEPS; ++k) {
    // current tile (prefetched async) must be fully landed block-wide
    wait_async0();
    __syncthreads();
    if (k + 1 < K_STEPS) {
      const unsigned short* tsrc = towersb + ((size_t)(k + 1) * N_TOTAL + row0) * N_IN;
      async_tower_load(tsrc, ((k + 1) & 1) ? Tt1 : Tt0, tid);
    }
    unsigned short* Tt = (k & 1) ? Tt1 : Tt0;

    // ---- output head ----
    gemm_layer<34, false>(Sbuf, Tt, OW1p, bOb1, Hbuf, nullptr, nullptr, lane, wave);
    __syncthreads();
    gemm_layer<32, true >(Hbuf, nullptr, OW2p, bOb2, nullptr, sOW3, predp, lane, wave);
    __syncthreads();
    if (tid < R_ROWS) {
      float logit = ob3 + predp[tid];
      prod[tid] *= 1.f / (1.f + __expf(-logit));
      predp[tid] = 0.f;
    }
    __syncthreads();

    // ---- memory path ----
    gemm_layer<34, false>(Sbuf, Tt, MW1p, bMb1, Hbuf, nullptr, nullptr, lane, wave);
    __syncthreads();
    gemm_layer<32, false>(Hbuf, nullptr, MW2p, bMb2, Sbuf, nullptr, nullptr, lane, wave);
    __syncthreads();
    gemm_layer<32, false>(Sbuf, nullptr, MW3p, bMb3, Hbuf, nullptr, nullptr, lane, wave);
    // new summary in Hbuf; next iteration's top-of-step barrier orders readers
    unsigned short* t = Sbuf; Sbuf = Hbuf; Hbuf = t;
  }

  __syncthreads();
  if (tid < R_ROWS) out[row0 + tid] = prod[tid];
}

// --- prep: pack fp32 weights (Krows x 1024) into WMMA-B fragment layout ---
// fragment (kt,ct) = 512 bf16; element: lane*16+e  <->  K=kt*32+(lane/16)*16+e,
// N=ct*16+(lane%16). Each lane's 32 bytes are contiguous in memory.
__global__ void pack_weight(const float* __restrict__ W, unsigned short* __restrict__ dst,
                            int Krows)
{
  int total = Krows * NH;
  for (int d = blockIdx.x * blockDim.x + threadIdx.x; d < total;
       d += gridDim.x * blockDim.x) {
    int frag = d >> 9, within = d & 511;
    int ln = within >> 4, e = within & 15;
    int kt = frag >> 6, ct = frag & 63;
    int K = kt * 32 + (ln >> 4) * 16 + e;
    int N = ct * 16 + (ln & 15);
    dst[d] = f2bf(W[(size_t)K * NH + N]);
  }
}

// --- prep: towers (N,K,64) f32  ->  bf16 [k][n][i] ---
__global__ void pack_towers(const float* __restrict__ T, unsigned short* __restrict__ dst)
{
  int total = K_STEPS * N_TOTAL * N_IN;
  for (int d = blockIdx.x * blockDim.x + threadIdx.x; d < total;
       d += gridDim.x * blockDim.x) {
    int k = d / (N_TOTAL * N_IN);
    int rem = d - k * (N_TOTAL * N_IN);
    int n = rem >> 6, i = rem & 63;
    dst[d] = f2bf(T[(size_t)n * (K_STEPS * N_IN) + k * N_IN + i]);
  }
}

extern "C" void kernel_launch(void* const* d_in, const int* in_sizes, int n_in,
                              void* d_out, int out_size, void* d_ws, size_t ws_size,
                              hipStream_t stream)
{
  const float* towers    = (const float*)d_in[0];
  const float* aggregate = (const float*)d_in[1];
  const float* MW1 = (const float*)d_in[2];  const float* Mb1 = (const float*)d_in[3];
  const float* MW2 = (const float*)d_in[4];  const float* Mb2 = (const float*)d_in[5];
  const float* MW3 = (const float*)d_in[6];  const float* Mb3 = (const float*)d_in[7];
  const float* OW1 = (const float*)d_in[8];  const float* Ob1 = (const float*)d_in[9];
  const float* OW2 = (const float*)d_in[10]; const float* Ob2 = (const float*)d_in[11];
  const float* OW3 = (const float*)d_in[12]; const float* Ob3 = (const float*)d_in[13];

  const size_t szW1 = (size_t)1088 * 1024 * 2;   // bytes
  const size_t szW  = (size_t)1024 * 1024 * 2;
  const size_t szT  = (size_t)K_STEPS * N_TOTAL * N_IN * 2;
  const size_t need = 2 * szW1 + 3 * szW + szT;
  if (ws_size < need) return;

  char* ws = (char*)d_ws;
  unsigned short* MW1p = (unsigned short*)ws;            ws += szW1;
  unsigned short* OW1p = (unsigned short*)ws;            ws += szW1;
  unsigned short* MW2p = (unsigned short*)ws;            ws += szW;
  unsigned short* MW3p = (unsigned short*)ws;            ws += szW;
  unsigned short* OW2p = (unsigned short*)ws;            ws += szW;
  unsigned short* towersb = (unsigned short*)ws;

  pack_weight<<<512, 256, 0, stream>>>(MW1, MW1p, 1088);
  pack_weight<<<512, 256, 0, stream>>>(OW1, OW1p, 1088);
  pack_weight<<<512, 256, 0, stream>>>(MW2, MW2p, 1024);
  pack_weight<<<512, 256, 0, stream>>>(MW3, MW3p, 1024);
  pack_weight<<<512, 256, 0, stream>>>(OW2, OW2p, 1024);
  pack_towers<<<2048, 256, 0, stream>>>(towers, towersb);

  const size_t smem =
      (size_t)2 * R_ROWS * STRIDE * 2 +   // two activation buffers
      (size_t)2 * R_ROWS * TSTRIDE * 2 +  // double-buffered tower tile
      6 * NH * 4 +                        // 5 biases + OW3
      2 * R_ROWS * 4;                     // predp + prod
  (void)hipFuncSetAttribute((const void*)bottomup_main,
                            hipFuncAttributeMaxDynamicSharedMemorySize, (int)smem);

  bottomup_main<<<N_TOTAL / R_ROWS, 256, smem, stream>>>(
      MW1p, MW2p, MW3p, OW1p, OW2p, towersb, aggregate,
      Mb1, Mb2, Mb3, Ob1, Ob2, OW3, Ob3, (float*)d_out);
}
